// KSSMBlock_44856638439974
// MI455X (gfx1250) — matrix operand order repaired and checked by reading, not compiled, not run on hardware
//
#include <hip/hip_runtime.h>
#include <hip/hip_bf16.h>

// ---------- problem constants ----------
#define BB   2
#define SS   2048
#define DM   1024
#define DI   2048
#define HH   32
#define HD   64
#define GG   32
#define NROW (BB*SS)        // 4096
#define P4DI (4*DI)         // 8192
#define NSMALL 352          // 96 dyn + 64 selB + 64 selC + 32 seldt + 32 beta + 32 rg + 32 ug

typedef __attribute__((ext_vector_type(16))) __bf16 v16bf;
typedef __attribute__((ext_vector_type(8)))  __bf16 bf16x8;
typedef __attribute__((ext_vector_type(8)))  float  v8f;

static __device__ __forceinline__ float sigmoidf_(float x) { return 1.f / (1.f + __expf(-x)); }
static __device__ __forceinline__ float softplusf_(float x) { return (x > 20.f) ? x : log1pf(__expf(x)); }
static __device__ __forceinline__ float siluf_(float x) { return x * sigmoidf_(x); }

// ---------- fp32 -> bf16 convert ----------
__global__ void kssm_cvt_bf16(const float* __restrict__ in, __bf16* __restrict__ out, size_t n) {
    size_t i = (size_t)blockIdx.x * blockDim.x + threadIdx.x;
    if (i < n) out[i] = (__bf16)in[i];
}

// ---------- RMSNorm + bf16 ----------
__global__ void kssm_rmsnorm(const float* __restrict__ x, const float* __restrict__ rms_w,
                             __bf16* __restrict__ xnbf) {
    int row = blockIdx.x;
    const float* xr = x + (size_t)row * DM;
    float s = 0.f;
    for (int i = threadIdx.x; i < DM; i += 256) { float v = xr[i]; s += v * v; }
    __shared__ float red[256];
    red[threadIdx.x] = s; __syncthreads();
    for (int off = 128; off > 0; off >>= 1) {
        if (threadIdx.x < off) red[threadIdx.x] += red[threadIdx.x + off];
        __syncthreads();
    }
    float scale = rsqrtf(red[0] / (float)DM + 1e-6f);
    for (int i = threadIdx.x; i < DM; i += 256)
        xnbf[(size_t)row * DM + i] = (__bf16)(xr[i] * scale * rms_w[i]);
}

// ---------- bf16 WMMA GEMM: C[M,N] = A[M,K] @ B[N,K]^T (+bias) (+resid) ----------
// K is a template constant so every fragment address is (32-bit voffset + immediate):
// no 64-bit address math and no pointer arrays in the hot loop.
// block = 256 threads (8 waves as 2x4); wave (wm,wn) computes a 32x64 tile.
// Fragments double-buffered (ping-pong) so global_load_b128 overlaps v_wmma.
// Requires M%64==0, N%256==0, K%64==0.
template <int K>
__global__ void __launch_bounds__(256, 1)
kssm_gemm_bf16(const __bf16* __restrict__ A, const __bf16* __restrict__ Bm,
               const float* __restrict__ bias, const float* __restrict__ resid,
               float* __restrict__ C, int M, int N) {
    int lane = threadIdx.x & 31;
    int wave = threadIdx.x >> 5;
    int wm = wave >> 2;                 // 0..1
    int wn = wave & 3;                  // 0..3
    int m0 = blockIdx.y * 64 + wm * 32;
    int n0 = blockIdx.x * 256 + wn * 64;
    int ml = lane & 15;
    int hi = lane >> 4;

    // 32-bit element offsets (buffers are < 2^31 bytes): saddr + voffset addressing.
    unsigned aoff = (unsigned)(m0 + ml) * (unsigned)K + (unsigned)(hi * 8);
    unsigned boff = (unsigned)(n0 + ml) * (unsigned)K + (unsigned)(hi * 8);

    v8f acc[2][4] = {};
    v16bf a0[2], b0[4], a1[2], b1[4];

    // A fragment (16-bit A 16x32): lane(m=ml,hi): elems 0..7 = K[hi*8..+7], 8..15 = K[16+hi*8..+7]
    // B fragment (16-bit B 32x16): lane(n=ml,hi): elems 0..7 = K[hi*8..+7], 8..15 = K[16+hi*8..+7]
    auto loadA = [&](v16bf (&aa)[2], unsigned off) {
        ((bf16x8*)&aa[0])[0] = *(const bf16x8*)(A + off);
        ((bf16x8*)&aa[0])[1] = *(const bf16x8*)(A + off + 16);
        ((bf16x8*)&aa[1])[0] = *(const bf16x8*)(A + off + 16 * K);
        ((bf16x8*)&aa[1])[1] = *(const bf16x8*)(A + off + 16 * K + 16);
    };
    auto loadB = [&](v16bf (&bb)[4], unsigned off) {
#pragma unroll
        for (int t = 0; t < 4; ++t) {
            ((bf16x8*)&bb[t])[0] = *(const bf16x8*)(Bm + off + t * 16 * K);
            ((bf16x8*)&bb[t])[1] = *(const bf16x8*)(Bm + off + t * 16 * K + 16);
        }
    };
    auto do_mma = [&](v16bf (&aa)[2], v16bf (&bb)[4]) {
#pragma unroll
        for (int t = 0; t < 4; ++t) {
            acc[0][t] = __builtin_amdgcn_wmma_f32_16x16x32_bf16(
                false, aa[0], false, bb[t], (short)0, acc[0][t], false, false);
            acc[1][t] = __builtin_amdgcn_wmma_f32_16x16x32_bf16(
                false, aa[1], false, bb[t], (short)0, acc[1][t], false, false);
        }
    };

    loadA(a0, aoff);
    loadB(b0, boff);
    for (int k0 = 0; k0 < K - 64; k0 += 64) {
        loadA(a1, aoff + 32);       // +32 elems = +64B: folds into instruction offset
        loadB(b1, boff + 32);
        do_mma(a0, b0);
        aoff += 64; boff += 64;
        loadA(a0, aoff);
        loadB(b0, boff);
        do_mma(a1, b1);
    }
    loadA(a1, aoff + 32);
    loadB(b1, boff + 32);
    do_mma(a0, b0);
    do_mma(a1, b1);

#pragma unroll
    for (int mi = 0; mi < 2; ++mi) {
#pragma unroll
        for (int t = 0; t < 4; ++t) {
            int n = n0 + t * 16 + ml;
            float bv = bias ? bias[n] : 0.f;
#pragma unroll
            for (int r = 0; r < 8; ++r) {
                int m = m0 + mi * 16 + r + hi * 8;  // C/D: vgpr r -> M=r (lanes0-15), r+8 (16-31)
                float v = acc[mi][t][r] + bv;
                if (resid) v += resid[(size_t)m * N + n];
                C[(size_t)m * N + n] = v;
            }
        }
    }
}

// ---------- causal depthwise conv(4) + SiLU -> u (f32 + bf16) ----------
__global__ void kssm_conv_silu(const float* __restrict__ proj, const float* __restrict__ conv_w,
                               const float* __restrict__ conv_b,
                               float* __restrict__ u, __bf16* __restrict__ ubf) {
    size_t idx = (size_t)blockIdx.x * blockDim.x + threadIdx.x;
    if (idx >= (size_t)NROW * DI) return;
    size_t row = idx >> 11;           // DI = 2048
    int c = (int)(idx & (DI - 1));
    int b = (int)(row >> 11);         // SS = 2048
    int s = (int)(row & (SS - 1));
    const float* w = conv_w + (size_t)c * 4;
    float acc = conv_b[c];
#pragma unroll
    for (int k = 0; k < 4; ++k) {
        int sp = s - 3 + k;
        if (sp >= 0)
            acc += proj[((size_t)(b * SS + sp)) * P4DI + 3 * DI + c] * w[k];
    }
    float uu = siluf_(acc);
    u[idx] = uu;
    ubf[idx] = (__bf16)uu;
}

// ---------- small projections: out[row][0..352) = u_row . W_row ----------
__global__ void kssm_small_proj(const float* __restrict__ u,
                                const float* __restrict__ dynW, const float* __restrict__ selBW,
                                const float* __restrict__ selCW, const float* __restrict__ seldtW,
                                const float* __restrict__ betaW, const float* __restrict__ rgW,
                                const float* __restrict__ ugW, float* __restrict__ out) {
    __shared__ float su[DI];
    int row = blockIdx.x;
    const float* ur = u + (size_t)row * DI;
    for (int i = threadIdx.x; i < DI; i += blockDim.x) su[i] = ur[i];
    __syncthreads();
    int t = threadIdx.x;
    if (t >= NSMALL) return;
    const float* wrow;
    if      (t <  96) wrow = dynW   + (size_t)(t      ) * DI;
    else if (t < 160) wrow = selBW  + (size_t)(t -  96) * DI;
    else if (t < 224) wrow = selCW  + (size_t)(t - 160) * DI;
    else if (t < 256) wrow = seldtW + (size_t)(t - 224) * DI;
    else if (t < 288) wrow = betaW  + (size_t)(t - 256) * DI;
    else if (t < 320) wrow = rgW    + (size_t)(t - 288) * DI;
    else              wrow = ugW    + (size_t)(t - 320) * DI;
    float s = 0.f;
    for (int i = 0; i < DI; ++i) s = fmaf(su[i], wrow[i], s);
    out[(size_t)row * NSMALL + t] = s;
}

// ---------- per-(b,s,h) scalar recurrence coefficients ----------
__global__ void kssm_scalars(const float* __restrict__ small, const float* __restrict__ dt_c,
                             const float* __restrict__ dyn_b, const float* __restrict__ beta_b,
                             const float* __restrict__ rg_b, const float* __restrict__ ug_b,
                             float* __restrict__ scal /* [N][H][8] */) {
    int idx = blockIdx.x * blockDim.x + threadIdx.x;
    if (idx >= NROW * HH) return;
    int row = idx >> 5, h = idx & (HH - 1);
    int s = row & (SS - 1);
    const float* sm = small + (size_t)row * NSMALL;
    float alpha = softplusf_(sm[h] + dyn_b[h]);
    float rope = __powf(10000.f, -(float)h / (float)HH);
    float omega = (sm[32 + h] + dyn_b[32 + h]) + (sm[64 + h] + dyn_b[64 + h]) + (float)s * rope;
    float dt = softplusf_(dt_c[h]) / (alpha + fabsf(omega) + 1e-4f);
    dt += softplusf_(sm[224 + h]);
    float a = 0.5f * dt * alpha;
    float w = 0.5f * dt * omega;
    float det = (1.f + a) * (1.f + a) + w * w;
    float lam2 = ((1.f - a) * (1.f - a) + w * w) / det;
    float r = sigmoidf_(sm[288 + h] + rg_b[h]);
    float vp = sqrtf(fmaxf(1.f - __powf(lam2, r), 1e-6f));
    float ug = sigmoidf_(sm[320 + h] + ug_b[h]);
    float bt = sigmoidf_(sm[256 + h] + beta_b[h]);
    float* o = scal + (size_t)idx * 8;
    o[0] = (1.f - a * a - w * w) / det;   // A11
    o[1] = 2.f * w / det;                 // A12
    o[2] = bt;
    o[3] = sm[96 + 2 * h];                // selB0
    o[4] = sm[96 + 2 * h + 1];            // selB1
    o[5] = sm[160 + 2 * h];               // selC0
    o[6] = sm[160 + 2 * h + 1];           // selC1
    o[7] = vp * ug;
}

// ---------- sequential scan over s; one lane per (b,h,d) chain ----------
__global__ void kssm_scan(const float* __restrict__ proj, const float* __restrict__ u,
                          const float* __restrict__ qp, const float* __restrict__ scal,
                          float* __restrict__ ypre) {
    int b = blockIdx.x >> 5;
    int h = blockIdx.x & (HH - 1);
    int d = threadIdx.x;           // 0..63
    int ch = h * HD + d;
    float S0 = 0.f, S1 = 0.f;
    for (int s = 0; s < SS; ++s) {
        size_t row = (size_t)b * SS + s;
        const float* sc = scal + (row * HH + h) * 8;
        float A11 = sc[0], A12 = sc[1], bt = sc[2];
        float sB0 = sc[3], sB1 = sc[4], sC0 = sc[5], sC1 = sc[6], vpug = sc[7];
        const float* pr = proj + row * P4DI;
        float k0 = pr[DI + 2 * ch] * sB0;
        float k1 = pr[DI + 2 * ch + 1] * sB1;
        float v = u[row * DI + ch] * vpug;
        float r0 = A11 * S0 + A12 * S1;
        float r1 = -A12 * S0 + A11 * S1;
        float err = v - (k0 * r0 + k1 * r1);
        S0 = r0 + bt * err * k0;
        S1 = r1 + bt * err * k1;
        float q0 = qp[row * (size_t)(2 * DI) + 2 * ch] * sC0;
        float q1 = qp[row * (size_t)(2 * DI) + 2 * ch + 1] * sC1;
        ypre[row * DI + ch] = q0 * S0 + q1 * S1;
    }
}

// ---------- groupnorm stats over (64 channels x 2048 positions) per (b,g) ----------
__global__ void kssm_gn_stats(const float* __restrict__ ypre, float* __restrict__ gnstat) {
    int b = blockIdx.x >> 5, g = blockIdx.x & (GG - 1);
    float sum = 0.f, sumsq = 0.f;
    for (int i = threadIdx.x; i < SS * 64; i += 256) {
        int s = i >> 6, c = g * 64 + (i & 63);
        float v = ypre[((size_t)(b * SS + s)) * DI + c];
        sum += v; sumsq += v * v;
    }
    __shared__ float rs[256], rq[256];
    rs[threadIdx.x] = sum; rq[threadIdx.x] = sumsq;
    __syncthreads();
    for (int off = 128; off > 0; off >>= 1) {
        if (threadIdx.x < off) { rs[threadIdx.x] += rs[threadIdx.x + off]; rq[threadIdx.x] += rq[threadIdx.x + off]; }
        __syncthreads();
    }
    if (threadIdx.x == 0) {
        float inv = 1.f / (float)(SS * 64);
        float mu = rs[0] * inv;
        float var = rq[0] * inv - mu * mu;
        gnstat[blockIdx.x * 2] = mu;
        gnstat[blockIdx.x * 2 + 1] = rsqrtf(var + 1e-5f);
    }
}

// ---------- groupnorm apply + silu(z) gate + D*u -> bf16 for out GEMM ----------
__global__ void kssm_finalize(const float* __restrict__ ypre, const float* __restrict__ gnstat,
                              const float* __restrict__ gn_w, const float* __restrict__ gn_b,
                              const float* __restrict__ proj, const float* __restrict__ u,
                              const float* __restrict__ Dp, __bf16* __restrict__ ybf) {
    size_t idx = (size_t)blockIdx.x * blockDim.x + threadIdx.x;
    if (idx >= (size_t)NROW * DI) return;
    size_t row = idx >> 11;
    int c = (int)(idx & (DI - 1));
    int b = (int)(row >> 11);
    int g = c >> 6;
    float mu = gnstat[(b * GG + g) * 2];
    float rstd = gnstat[(b * GG + g) * 2 + 1];
    float yn = (ypre[idx] - mu) * rstd * gn_w[c] + gn_b[c];
    float z = proj[row * P4DI + c];
    float y = yn * siluf_(z) + Dp[c] * u[idx];
    ybf[idx] = (__bf16)y;
}

extern "C" void kernel_launch(void* const* d_in, const int* in_sizes, int n_in,
                              void* d_out, int out_size, void* d_ws, size_t ws_size,
                              hipStream_t stream) {
    const float* x        = (const float*)d_in[0];
    const float* in_projW = (const float*)d_in[1];
    const float* in_projb = (const float*)d_in[2];
    const float* conv_w   = (const float*)d_in[3];
    const float* conv_b   = (const float*)d_in[4];
    const float* dyn_W    = (const float*)d_in[5];
    const float* dyn_b    = (const float*)d_in[6];
    const float* dt_c     = (const float*)d_in[7];
    const float* selB_W   = (const float*)d_in[8];
    const float* selC_W   = (const float*)d_in[9];
    const float* seldt_W  = (const float*)d_in[10];
    const float* beta_W   = (const float*)d_in[11];
    const float* beta_b   = (const float*)d_in[12];
    const float* rg_W     = (const float*)d_in[13];
    const float* rg_b     = (const float*)d_in[14];
    const float* ug_W     = (const float*)d_in[15];
    const float* ug_b     = (const float*)d_in[16];
    const float* Q_W      = (const float*)d_in[17];
    const float* out_W    = (const float*)d_in[18];
    const float* Dp       = (const float*)d_in[19];
    const float* rms_w    = (const float*)d_in[20];
    const float* gn_w     = (const float*)d_in[21];
    const float* gn_b     = (const float*)d_in[22];
    float* out = (float*)d_out;

    char* ws = (char*)d_ws;
    size_t off = 0;
    auto alloc = [&](size_t bytes) { size_t o = off; off += (bytes + 255) & ~(size_t)255; return o; };

    __bf16* xnbf  = (__bf16*)(ws + alloc((size_t)NROW * DM * 2));
    __bf16* wipbf = (__bf16*)(ws + alloc((size_t)P4DI * DM * 2));
    __bf16* wqbf  = (__bf16*)(ws + alloc((size_t)(2 * DI) * DI * 2));
    __bf16* wobf  = (__bf16*)(ws + alloc((size_t)DM * DI * 2));
    float*  proj  = (float*)(ws + alloc((size_t)NROW * P4DI * 4));
    float*  u     = (float*)(ws + alloc((size_t)NROW * DI * 4));
    __bf16* ubf   = (__bf16*)(ws + alloc((size_t)NROW * DI * 2));
    float*  qp    = (float*)(ws + alloc((size_t)NROW * (2 * DI) * 4));
    float*  small = (float*)(ws + alloc((size_t)NROW * NSMALL * 4));
    float*  scal  = (float*)(ws + alloc((size_t)NROW * HH * 8 * 4));
    float*  ypre  = (float*)(ws + alloc((size_t)NROW * DI * 4));
    __bf16* ybf   = (__bf16*)(ws + alloc((size_t)NROW * DI * 2));
    float*  gnst  = (float*)(ws + alloc((size_t)BB * GG * 2 * 4));
    (void)ws_size; (void)in_sizes; (void)n_in; (void)out_size;

    // 1. RMSNorm -> bf16
    kssm_rmsnorm<<<NROW, 256, 0, stream>>>(x, rms_w, xnbf);

    // 2. weight conversions fp32 -> bf16
    {
        size_t n1 = (size_t)P4DI * DM;
        kssm_cvt_bf16<<<(unsigned)((n1 + 255) / 256), 256, 0, stream>>>(in_projW, wipbf, n1);
        size_t n2 = (size_t)(2 * DI) * DI;
        kssm_cvt_bf16<<<(unsigned)((n2 + 255) / 256), 256, 0, stream>>>(Q_W, wqbf, n2);
        size_t n3 = (size_t)DM * DI;
        kssm_cvt_bf16<<<(unsigned)((n3 + 255) / 256), 256, 0, stream>>>(out_W, wobf, n3);
    }

    // 3. in_proj GEMM: proj[N x 8192] = xn @ in_proj_W^T + b   (K = DM = 1024)
    kssm_gemm_bf16<DM><<<dim3(P4DI / 256, NROW / 64), 256, 0, stream>>>(
        xnbf, wipbf, in_projb, nullptr, proj, NROW, P4DI);

    // 4. depthwise causal conv + SiLU -> u
    {
        size_t n = (size_t)NROW * DI;
        kssm_conv_silu<<<(unsigned)((n + 255) / 256), 256, 0, stream>>>(proj, conv_w, conv_b, u, ubf);
    }

    // 5. Q GEMM: qp[N x 4096] = u @ Q_W^T   (K = DI = 2048)
    kssm_gemm_bf16<DI><<<dim3((2 * DI) / 256, NROW / 64), 256, 0, stream>>>(
        ubf, wqbf, nullptr, nullptr, qp, NROW, 2 * DI);

    // 6. small projections
    kssm_small_proj<<<NROW, 384, 0, stream>>>(u, dyn_W, selB_W, selC_W, seldt_W, beta_W, rg_W, ug_W, small);

    // 7. per-(b,s,h) scalars
    kssm_scalars<<<(NROW * HH + 255) / 256, 256, 0, stream>>>(small, dt_c, dyn_b, beta_b, rg_b, ug_b, scal);

    // 8. sequential scan
    kssm_scan<<<BB * HH, HD, 0, stream>>>(proj, u, qp, scal, ypre);

    // 9. groupnorm stats
    kssm_gn_stats<<<BB * GG, 256, 0, stream>>>(ypre, gnst);

    // 10. groupnorm apply + gates -> bf16
    {
        size_t n = (size_t)NROW * DI;
        kssm_finalize<<<(unsigned)((n + 255) / 256), 256, 0, stream>>>(ypre, gnst, gn_w, gn_b, proj, u, Dp, ybf);
    }

    // 11. out GEMM + residual: out[N x 1024] = x + ybf @ out_W^T   (K = DI = 2048)
    kssm_gemm_bf16<DI><<<dim3(DM / 256, NROW / 64), 256, 0, stream>>>(
        ybf, wobf, nullptr, x, out, NROW, DM);
}
